// Attention_52750788329629
// MI455X (gfx1250) — 16-bit and 8-bit matrix operand order repaired and checked by reading, compiled, not run on hardware
//
#include <hip/hip_runtime.h>

#define DIM    1024
#define NH     16
#define HD     64
#define GRP    128
#define NB     2
#define TT     2048
#define NT     (NB*TT)
#define KSTEPS (DIM/64)
#define NGRP   (DIM/GRP)
#define NTILE  (DIM/16)

typedef __attribute__((ext_vector_type(16))) _Float16 v16h;
typedef __attribute__((ext_vector_type(8)))  _Float16 v8h;
typedef __attribute__((ext_vector_type(8)))  float    v8f;
typedef __attribute__((ext_vector_type(8)))  int      v8i;
typedef __attribute__((ext_vector_type(4)))  int      v4i;

__global__ void quant_pack_weights(const float* __restrict__ w0,
                                   const float* __restrict__ w1,
                                   const float* __restrict__ w2,
                                   const float* __restrict__ w3,
                                   signed char* __restrict__ packW,
                                   float* __restrict__ wsW) {
  const int mat = blockIdx.y;
  const float* w = (mat == 0) ? w0 : (mat == 1) ? w1 : (mat == 2) ? w2 : w3;
  const int o = blockIdx.x / NGRP;
  const int g = blockIdx.x % NGRP;
  const int t = threadIdx.x;
  __shared__ float red[128];
  const float val = w[(size_t)o * DIM + g * GRP + t];
  red[t] = fabsf(val);
  __syncthreads();
  for (int s = 64; s > 0; s >>= 1) {
    if (t < s) red[t] += red[t + s];
    __syncthreads();
  }
  const float ws = red[0] / (float)GRP + 1e-5f;
  if (t == 0) wsW[((size_t)mat * DIM + o) * NGRP + g] = ws;
  float q = rintf(val / ws);
  q = fminf(1.f, fmaxf(-1.f, q));
  const signed char qb = (signed char)(int)q;
  const int k = g * GRP + t;
  const int kstep = k >> 6;
  const int K = k & 63;
  const int N = o & 15;
  const int nt = o >> 4;
  const int lane = N + 16 * ((K >> 3) & 1);
  const int dw = (K >> 4) * 2 + ((K & 7) >> 2);
  const int off = lane * 32 + dw * 4 + (K & 3);
  packW[(((size_t)mat * NTILE + nt) * KSTEPS + kstep) * 1024 + off] = qb;
}

__global__ void quant_pack_acts(const float* __restrict__ x,
                                float* __restrict__ invs,
                                signed char* __restrict__ packA) {
  const int tok = blockIdx.x;
  const int tid = threadIdx.x;
  const float* xr = x + (size_t)tok * DIM;
  __shared__ float red[256];
  float m = 0.f;
  for (int i = tid; i < DIM; i += 256) m = fmaxf(m, fabsf(xr[i]));
  red[tid] = m;
  __syncthreads();
  for (int s = 128; s > 0; s >>= 1) {
    if (tid < s) red[tid] = fmaxf(red[tid], red[tid + s]);
    __syncthreads();
  }
  const float s = 127.f / (red[0] + 1e-5f);
  if (tid == 0) invs[tok] = 1.f / s;
  const int mt = tok >> 4, mm = tok & 15;
  const size_t base = (size_t)mt * KSTEPS * 1024;
  for (int k = tid; k < DIM; k += 256) {
    float q = rintf(xr[k] * s);
    q = fminf(127.f, fmaxf(-128.f, q));
    const int kstep = k >> 6;
    const int K = k & 63;
    const int lane = mm + 16 * ((K >> 3) & 1);
    const int chunk = K >> 4;
    const int off = lane * 32 + chunk * 8 + (K & 7);
    packA[base + (size_t)kstep * 1024 + off] = (signed char)(int)q;
  }
}

template <bool F16OUT>
__global__ void bitlinear_gemm(const signed char* __restrict__ packA,
                               const float* __restrict__ invs,
                               const signed char* __restrict__ packW,
                               const float* __restrict__ wsW,
                               _Float16* __restrict__ outH,
                               float* __restrict__ outF) {
  __shared__ __align__(32) signed char Als[KSTEPS * 1024];
  const int mt = blockIdx.y;
  const int lane = threadIdx.x & 31;
  const int wv = threadIdx.x >> 5;
  const int nt = blockIdx.x * 8 + wv;

  const signed char* Ag = packA + (size_t)mt * KSTEPS * 1024;
  for (int i = threadIdx.x * 16; i < KSTEPS * 1024; i += 256 * 16)
    *(v4i*)(Als + i) = *(const v4i*)(Ag + i);
  __syncthreads();

  const signed char* Wg = packW + (size_t)nt * KSTEPS * 1024;
  v8f accf = {};
  const int ocol = nt * 16 + (lane & 15);
#pragma unroll
  for (int g = 0; g < NGRP; ++g) {
    v8i acc = {};
#pragma unroll
    for (int kk = 0; kk < 2; ++kk) {
      const int ks = g * 2 + kk;
      __builtin_prefetch(Wg + (size_t)(ks + 2 < KSTEPS ? ks + 2 : ks) * 1024 + lane * 32, 0, 1);
      v8i a = *(const v8i*)(Als + ks * 1024 + lane * 32);
      v8i b = *(const v8i*)(Wg + (size_t)ks * 1024 + lane * 32);
      acc = __builtin_amdgcn_wmma_i32_16x16x64_iu8(true, a, true, b, acc, false, false);
    }
    const float ws = wsW[(size_t)ocol * NGRP + g];
#pragma unroll
    for (int i = 0; i < 8; ++i) accf[i] += ws * (float)acc[i];
  }
  const int mrow0 = mt * 16 + 8 * (lane >> 4);
#pragma unroll
  for (int i = 0; i < 8; ++i) {
    const float y = accf[i] * invs[mrow0 + i];
    const size_t idx = (size_t)(mrow0 + i) * DIM + ocol;
    if constexpr (F16OUT) outH[idx] = (_Float16)y;
    else                  outF[idx] = y;
  }
}

__global__ void rope_convert(const _Float16* __restrict__ Qg,
                             const _Float16* __restrict__ Kg,
                             const _Float16* __restrict__ Vg,
                             const float* __restrict__ cosT,
                             const float* __restrict__ sinT,
                             _Float16* __restrict__ Qh,
                             _Float16* __restrict__ Kt,
                             _Float16* __restrict__ Vh) {
  const int tok = blockIdx.x;
  const int b = tok / TT, t = tok % TT;
  for (int p = threadIdx.x; p < DIM / 2; p += 256) {
    const int h = p / 32;
    const int j = p % 32;
    const float c = cosT[t * 32 + j], s = sinT[t * 32 + j];
    const size_t src = (size_t)tok * DIM + h * HD + 2 * j;
    const float qe = (float)Qg[src], qo = (float)Qg[src + 1];
    const float ke = (float)Kg[src], ko = (float)Kg[src + 1];
    const size_t hb = (((size_t)b * NH + h) * TT + t) * HD;
    Qh[hb + 2 * j]     = (_Float16)(qe * c - qo * s);
    Qh[hb + 2 * j + 1] = (_Float16)(qe * s + qo * c);
    const size_t ktb = (((size_t)b * NH + h) * HD) * TT;
    Kt[ktb + (size_t)(2 * j) * TT + t]     = (_Float16)(ke * c - ko * s);
    Kt[ktb + (size_t)(2 * j + 1) * TT + t] = (_Float16)(ke * s + ko * c);
    Vh[hb + 2 * j]     = Vg[src];
    Vh[hb + 2 * j + 1] = Vg[src + 1];
  }
}

__global__ void flash_attn(const _Float16* __restrict__ Qh,
                           const _Float16* __restrict__ Kt,
                           const _Float16* __restrict__ Vh,
                           float* __restrict__ attnout) {
  const int blk = blockIdx.x;
  const int qblk = blk % (TT / 128);
  const int bh = blk / (TT / 128);
  const int b = bh / NH, h = bh % NH;
  const int lane = threadIdx.x & 31, wv = threadIdx.x >> 5;
  const int m = lane & 15, hl = lane >> 4;
  const int qbase = qblk * 128 + wv * 16;

  __shared__ __align__(32) _Float16 KF[4 * 32 * 16];
  __shared__ __align__(32) _Float16 VF[4 * 32 * 16];
  __shared__ __align__(32) _Float16 PF[8 * 32 * 16];

  const _Float16* Kb = Kt + (size_t)bh * HD * TT;
  const _Float16* Vb = Vh + (size_t)bh * TT * HD;
  const _Float16* Qb = Qh + (size_t)bh * TT * HD;

  v16h aq0, aq1;
  {
    const _Float16* qrow = Qb + (size_t)(qbase + m) * HD;
    v8h l0 = *(const v8h*)(qrow + hl * 8);
    v8h h0 = *(const v8h*)(qrow + 16 + hl * 8);
    v8h l1 = *(const v8h*)(qrow + 32 + hl * 8);
    v8h h1 = *(const v8h*)(qrow + 48 + hl * 8);
#pragma unroll
    for (int i = 0; i < 8; ++i) {
      aq0[i] = l0[i]; aq0[i + 8] = h0[i];
      aq1[i] = l1[i]; aq1[i + 8] = h1[i];
    }
  }

  float mrun[8], lrun[8];
  v8f O[4];
#pragma unroll
  for (int i = 0; i < 8; ++i) { mrun[i] = -1e30f; lrun[i] = 0.f; }
#pragma unroll
  for (int dc = 0; dc < 4; ++dc) O[dc] = (v8f){};

  const int kblocks = qblk * 4 + 4;
  for (int kb = 0; kb < kblocks; ++kb) {
    const int k0 = kb * 32;

    for (int i = threadIdx.x; i < HD * 4; i += 256) {
      const int d = i >> 2, c = i & 3;
      const v8h kv = *(const v8h*)(Kb + (size_t)d * TT + k0 + c * 8);
      const int dcF = d >> 5;
      const int dl = d & 31;
      const int hh = (dl & 7) + ((dl >> 4) << 3);
      const int lanehi = ((dl >> 3) & 1) << 4;
#pragma unroll
      for (int j = 0; j < 8; ++j) {
        const int key = c * 8 + j;
        const int f = dcF * 2 + (key >> 4);
        KF[f * 512 + ((key & 15) + lanehi) * 16 + hh] = kv[j];
      }
    }
    for (int i = threadIdx.x; i < 32 * 8; i += 256) {
      const int key = i >> 3, c = i & 7;
      const v8h vvv = *(const v8h*)(Vb + (size_t)(k0 + key) * HD + c * 8);
      const int hh = (key & 7) + ((key >> 4) << 3);
      const int lanehi = ((key >> 3) & 1) << 4;
#pragma unroll
      for (int j = 0; j < 8; ++j) {
        const int d = c * 8 + j;
        VF[(d >> 4) * 512 + ((d & 15) + lanehi) * 16 + hh] = vvv[j];
      }
    }
    __syncthreads();

    const bool active = (qbase + 15) >= k0;
    if (active) {
      const v16h b00 = *(const v16h*)(KF + 0 * 512 + lane * 16);
      const v16h b10 = *(const v16h*)(KF + 1 * 512 + lane * 16);
      const v16h b01 = *(const v16h*)(KF + 2 * 512 + lane * 16);
      const v16h b11 = *(const v16h*)(KF + 3 * 512 + lane * 16);
      v8f S0 = {}, S1 = {};
      S0 = __builtin_amdgcn_wmma_f32_16x16x32_f16(false, aq0, false, b00, (short)0, S0, false, false);
      S0 = __builtin_amdgcn_wmma_f32_16x16x32_f16(false, aq1, false, b01, (short)0, S0, false, false);
      S1 = __builtin_amdgcn_wmma_f32_16x16x32_f16(false, aq0, false, b10, (short)0, S1, false, false);
      S1 = __builtin_amdgcn_wmma_f32_16x16x32_f16(false, aq1, false, b11, (short)0, S1, false, false);

      const float sc = 0.125f;
      const int qr0 = qbase + 8 * hl;
      float rmax[8];
#pragma unroll
      for (int i = 0; i < 8; ++i) {
        const int q = qr0 + i;
        const float s0 = (k0 + m      <= q) ? S0[i] * sc : -1e30f;
        const float s1 = (k0 + 16 + m <= q) ? S1[i] * sc : -1e30f;
        S0[i] = s0; S1[i] = s1;
        rmax[i] = fmaxf(s0, s1);
      }
#pragma unroll
      for (int off = 1; off < 16; off <<= 1) {
#pragma unroll
        for (int i = 0; i < 8; ++i)
          rmax[i] = fmaxf(rmax[i], __shfl_xor(rmax[i], off, 32));
      }
      float corr[8];
#pragma unroll
      for (int i = 0; i < 8; ++i) {
        const float nm = fmaxf(mrun[i], rmax[i]);
        corr[i] = __expf(mrun[i] - nm);
        mrun[i] = nm;
        S0[i] = __expf(S0[i] - nm);
        S1[i] = __expf(S1[i] - nm);
      }
#pragma unroll
      for (int i = 0; i < 8; ++i) {
        float ps = S0[i] + S1[i];
#pragma unroll
        for (int off = 1; off < 16; off <<= 1) ps += __shfl_xor(ps, off, 32);
        lrun[i] = lrun[i] * corr[i] + ps;
#pragma unroll
        for (int dc = 0; dc < 4; ++dc) O[dc][i] *= corr[i];
      }
      const int pb = wv * 512;
      const int lhi = (m >> 3) << 4;
      const int hh0 = m & 7;
#pragma unroll
      for (int i = 0; i < 8; ++i) {
        const int r = 8 * hl + i;
        PF[pb + (r + lhi) * 16 + hh0]     = (_Float16)S0[i];
        PF[pb + (r + lhi) * 16 + 8 + hh0] = (_Float16)S1[i];
      }
    }
    __syncthreads();
    if (active) {
      const v16h ap = *(const v16h*)(PF + wv * 512 + lane * 16);
#pragma unroll
      for (int dc = 0; dc < 4; ++dc) {
        const v16h bv = *(const v16h*)(VF + dc * 512 + lane * 16);
        O[dc] = __builtin_amdgcn_wmma_f32_16x16x32_f16(false, ap, false, bv, (short)0, O[dc], false, false);
      }
    }
    __syncthreads();
  }

#pragma unroll
  for (int i = 0; i < 8; ++i) {
    const int q = qbase + 8 * hl + i;
    const float inv = 1.f / lrun[i];
#pragma unroll
    for (int dc = 0; dc < 4; ++dc)
      attnout[((size_t)(b * TT + q)) * DIM + h * HD + dc * 16 + m] = O[dc][i] * inv;
  }
}

extern "C" void kernel_launch(void* const* d_in, const int* in_sizes, int n_in,
                              void* d_out, int out_size, void* d_ws, size_t ws_size,
                              hipStream_t stream) {
  (void)in_sizes; (void)n_in; (void)out_size; (void)ws_size;
  const float* x    = (const float*)d_in[0];
  const float* cosT = (const float*)d_in[1];
  const float* sinT = (const float*)d_in[2];
  const float* wq   = (const float*)d_in[3];
  const float* wk   = (const float*)d_in[4];
  const float* wv   = (const float*)d_in[5];
  const float* wo   = (const float*)d_in[6];
  float* out = (float*)d_out;

  char* ws = (char*)d_ws;
  size_t off = 0;
  auto alloc = [&](size_t bytes) {
    size_t r = off;
    off = (off + bytes + 255) & ~(size_t)255;
    return r;
  };
  const size_t PACKW_BYTES = (size_t)NTILE * KSTEPS * 1024;
  signed char* packW  = (signed char*)(ws + alloc(4 * PACKW_BYTES));
  float*       wsW    = (float*)(ws + alloc((size_t)4 * DIM * NGRP * 4));
  float*       invsX  = (float*)(ws + alloc((size_t)NT * 4));
  signed char* packAx = (signed char*)(ws + alloc((size_t)NT * DIM));
  _Float16*    Qg     = (_Float16*)(ws + alloc((size_t)NT * DIM * 2));
  _Float16*    Kg     = (_Float16*)(ws + alloc((size_t)NT * DIM * 2));
  _Float16*    Vg     = (_Float16*)(ws + alloc((size_t)NT * DIM * 2));
  _Float16*    Qh     = (_Float16*)(ws + alloc((size_t)NT * DIM * 2));
  _Float16*    KtB    = (_Float16*)(ws + alloc((size_t)NT * DIM * 2));
  _Float16*    Vh     = (_Float16*)(ws + alloc((size_t)NT * DIM * 2));
  float*       attnO  = (float*)(ws + alloc((size_t)NT * DIM * 4));
  float*       invsA  = (float*)(ws + alloc((size_t)NT * 4));
  signed char* packAa = (signed char*)(ws + alloc((size_t)NT * DIM));

  quant_pack_weights<<<dim3(DIM * NGRP, 4), 128, 0, stream>>>(wq, wk, wv, wo, packW, wsW);
  quant_pack_acts<<<NT, 256, 0, stream>>>(x, invsX, packAx);
  const dim3 gGrid(DIM / 128, NT / 16);
  bitlinear_gemm<true><<<gGrid, 256, 0, stream>>>(packAx, invsX, packW + 0 * PACKW_BYTES,
                                                  wsW + 0 * (size_t)DIM * NGRP, Qg, nullptr);
  bitlinear_gemm<true><<<gGrid, 256, 0, stream>>>(packAx, invsX, packW + 1 * PACKW_BYTES,
                                                  wsW + 1 * (size_t)DIM * NGRP, Kg, nullptr);
  bitlinear_gemm<true><<<gGrid, 256, 0, stream>>>(packAx, invsX, packW + 2 * PACKW_BYTES,
                                                  wsW + 2 * (size_t)DIM * NGRP, Vg, nullptr);
  rope_convert<<<NT, 256, 0, stream>>>(Qg, Kg, Vg, cosT, sinT, Qh, KtB, Vh);
  flash_attn<<<(NB * NH) * (TT / 128), 256, 0, stream>>>(Qh, KtB, Vh, attnO);
  quant_pack_acts<<<NT, 256, 0, stream>>>(attnO, invsA, packAa);
  bitlinear_gemm<false><<<gGrid, 256, 0, stream>>>(packAa, invsA, packW + 3 * PACKW_BYTES,
                                                   wsW + 3 * (size_t)DIM * NGRP, nullptr, out);
}
